// AdaptiveLSTM_1374389534805
// MI455X (gfx1250) — matrix operand order repaired and checked by reading, not compiled, not run on hardware
//
#include <hip/hip_runtime.h>
#include <hip/hip_bf16.h>

// ---------------------------------------------------------------------------
// AdaptiveLSTM (2-layer, modality-add) for MI455X / gfx1250, wave32 + WMMA.
// B=64, S=512, D_IN=512, H=1024, D_ADD=128, OUT=3.
//   * bf16 data / fp32 accumulate via v_wmma_f32_16x16x32_bf16.
//   * Weights transposed to [N,K] bf16 (contiguous WMMA operand loads).
//   * Time-parallel projections P0/P1 precomputed (fp32), used as WMMA C-init.
//   * Recurrence: block = one 16-wide hidden tile, 4 waves = 4 batch tiles.
//     Gate weights staged into LDS with GLOBAL_LOAD_ASYNC_TO_LDS_B128
//     (ASYNCcnt, double-buffered) so the 4 waves share each weight byte and
//     L2 traffic drops 4x; B operands read from LDS as ds_load_b128.
// ---------------------------------------------------------------------------

typedef __bf16 bf16_t;
typedef __attribute__((ext_vector_type(8)))  __bf16 v8bf;
typedef __attribute__((ext_vector_type(16))) __bf16 v16bf;
typedef __attribute__((ext_vector_type(8)))  float  v8f;

#define B_SZ   64
#define S_LEN  512
#define DIN    512
#define H_DIM  1024
#define DADD   128
#define N4     4096   // 4*H
#define OUTD   3
#define MTOT   (B_SZ * S_LEN)   // 32768 flattened (b,s) rows
#define NKCH   (H_DIM / 32)     // 32 K-chunks in the recurrent GEMMs

__device__ __forceinline__ bf16_t f2bf(float f) {
  union { float f; unsigned u; } v; v.f = f;
  unsigned r = v.u + 0x7FFFu + ((v.u >> 16) & 1u);   // round-to-nearest-even
  unsigned short h = (unsigned short)(r >> 16);
  return __builtin_bit_cast(bf16_t, h);
}

__device__ __forceinline__ v8f wmma_bf16(v16bf a, v16bf b, v8f c) {
  return __builtin_amdgcn_wmma_f32_16x16x32_bf16(false, a, false, b, (short)0, c,
                                                 false, false);
}

__device__ __forceinline__ v16bf cat16(v8bf lo, v8bf hi) {
  return __builtin_shufflevector(lo, hi, 0, 1, 2, 3, 4, 5, 6, 7,
                                         8, 9, 10, 11, 12, 13, 14, 15);
}

// A tile: 16x32 bf16 of row-major A at (row0, k0), stride lda.
// lane<16 -> M=lane, K={kb..kb+15}, kb = (lane>>4)*16.
__device__ __forceinline__ v16bf load_a_tile(const bf16_t* __restrict__ A, int lda,
                                             int row0, int k0, int lane) {
  int lm = lane & 15, lh = lane >> 4;
  const bf16_t* ap = A + (size_t)(row0 + lm) * lda + k0 + lh * 16;
  v8bf lo = *(const v8bf*)ap;
  v8bf hi = *(const v8bf*)(ap + 8);
  return cat16(lo, hi);
}

// B tile (global path, used in precompute): Bt stored [N,K]; lane=N in half,
// per-lane K contiguous (koff = (lane>>4)*16).
__device__ __forceinline__ v16bf load_b_tile(const bf16_t* __restrict__ Bt, int K,
                                             int col0, int k0, int lane) {
  int lm = lane & 15, lh = lane >> 4;
  const bf16_t* bp = Bt + (size_t)(col0 + lm) * K + k0 + lh * 16;
  v8bf lo = *(const v8bf*)bp;
  v8bf hi = *(const v8bf*)(bp + 8);
  return cat16(lo, hi);
}

__device__ __forceinline__ float sigm(float x) { return 1.0f / (1.0f + __expf(-x)); }

// -------------------- conversion kernels -----------------------------------

// fp32 [K,N] row-major  ->  bf16 [N,K] (transposed), 32x32 LDS tiles.
__global__ void k_cvt_t(const float* __restrict__ src, bf16_t* __restrict__ dst,
                        int K, int N) {
  __shared__ bf16_t tile[32][33];
  int nt = blockIdx.x * 32;
  int kt = blockIdx.y * 32;
  int tx = threadIdx.x & 31, ty = threadIdx.x >> 5;
  for (int r = ty; r < 32; r += 8)
    tile[r][tx] = f2bf(src[(size_t)(kt + r) * N + nt + tx]);
  __syncthreads();
  for (int r = ty; r < 32; r += 8)
    dst[(size_t)(nt + r) * K + kt + tx] = tile[tx][r];
}

__global__ void k_cvt(const float* __restrict__ src, bf16_t* __restrict__ dst,
                      size_t n) {
  size_t i = (size_t)blockIdx.x * blockDim.x + threadIdx.x;
  if (i < n) dst[i] = f2bf(src[i]);
}

// -------------------- time-parallel projection GEMM ------------------------
// P[Mtot, 4096] = A1 @ B1t^T (+ A2 @ B2t^T) + bias1 (+ bias2).
// Wave computes 16 rows x 64 cols (4 accs). Grid exact -> EXEC all 1s.
__global__ void __launch_bounds__(128)
k_precompute(const bf16_t* __restrict__ A1, int lda1, int K1,
             const bf16_t* __restrict__ B1t,
             const bf16_t* __restrict__ A2, int lda2, int K2,
             const bf16_t* __restrict__ B2t,
             const float* __restrict__ bias1, const float* __restrict__ bias2,
             float* __restrict__ P) {
  int lane = threadIdx.x & 31;
  int widx = blockIdx.x * 4 + (threadIdx.x >> 5);
  const int ntiles = N4 / 64;
  int mt = widx / ntiles, nt = widx - mt * ntiles;
  int row0 = mt * 16, col0 = nt * 64;
  int lm = lane & 15, lh = lane >> 4;

  v8f acc[4];
#pragma unroll
  for (int j = 0; j < 4; ++j) {
    int col = col0 + j * 16 + lm;
    float bsum = bias1[col] + (bias2 ? bias2[col] : 0.0f);
#pragma unroll
    for (int r = 0; r < 8; ++r) acc[j][r] = bsum;
  }
  for (int k0 = 0; k0 < K1; k0 += 32) {
    v16bf av = load_a_tile(A1, lda1, row0, k0, lane);
#pragma unroll
    for (int j = 0; j < 4; ++j) {
      v16bf bv = load_b_tile(B1t, K1, col0 + j * 16, k0, lane);
      acc[j] = wmma_bf16(av, bv, acc[j]);
    }
  }
  if (A2) {
    for (int k0 = 0; k0 < K2; k0 += 32) {
      v16bf av = load_a_tile(A2, lda2, row0, k0, lane);
#pragma unroll
      for (int j = 0; j < 4; ++j) {
        v16bf bv = load_b_tile(B2t, K2, col0 + j * 16, k0, lane);
        acc[j] = wmma_bf16(av, bv, acc[j]);
      }
    }
  }
#pragma unroll
  for (int j = 0; j < 4; ++j)
#pragma unroll
    for (int r = 0; r < 8; ++r) {
      int row = row0 + lh * 8 + r;
      int col = col0 + j * 16 + lm;
      P[(size_t)row * N4 + col] = acc[j][r];
    }
}

// -------------------- async-to-LDS weight staging ---------------------------
// Stage 4KB = {4 gates x 16 cols} x {32 K} bf16 for one K-chunk into LDS.
// Wave w stages gate g=w (16 rows x 64B = 64 x 16B chunks -> 2 async ops/wave).
__device__ __forceinline__ void issue_b_stage(const bf16_t* __restrict__ Bt,
                                              int col0, int k0, int buf,
                                              int w, int lane, unsigned ldsBase) {
#pragma unroll
  for (int i = 0; i < 2; ++i) {
    int chunk = i * 32 + lane;            // 0..63 within this wave's gate
    int colL  = chunk >> 2;               // 0..15
    int c4    = chunk & 3;                // 16B sub-chunk within 64B row
    const bf16_t* gp = Bt + (size_t)(w * H_DIM + col0 + colL) * H_DIM + k0;
    unsigned long long gaddr = (unsigned long long)(const void*)gp +
                               (unsigned long long)(c4 * 16);
    unsigned loff = ldsBase + (unsigned)(buf * 4096 + (w * 16 + colL) * 64 + c4 * 16);
    asm volatile("global_load_async_to_lds_b128 %0, %1, off"
                 :: "v"(loff), "v"(gaddr) : "memory");
  }
}

// One recurrent GEMM phase: acc[g] += A[16x1024 tile] @ Bt[gate g cols].
// bsm: [2][4 gates][16 cols][32 K] bf16, 16B aligned.
__device__ __forceinline__ void gemm_phase(const bf16_t* __restrict__ A,
                                           const bf16_t* __restrict__ Bt,
                                           bf16_t (*bsm)[4][16][32],
                                           unsigned ldsBase,
                                           int row0, int col0, int w, int lane,
                                           v8f acc[4]) {
  int lm = lane & 15, lh = lane >> 4;
  issue_b_stage(Bt, col0, 0, 0, w, lane, ldsBase);
#pragma unroll 2
  for (int kc = 0; kc < NKCH; ++kc) {
    int cur = kc & 1;
    __syncthreads();                       // buf[cur^1] free to overwrite
    if (kc + 1 < NKCH) {
      issue_b_stage(Bt, col0, (kc + 1) * 32, cur ^ 1, w, lane, ldsBase);
      asm volatile("s_wait_asynccnt 0x2" ::: "memory");   // oldest 2 (cur) done
    } else {
      asm volatile("s_wait_asynccnt 0x0" ::: "memory");
    }
    __syncthreads();                       // all waves' cur stage visible
    v16bf av = load_a_tile(A, H_DIM, row0, kc * 32, lane);
#pragma unroll
    for (int g = 0; g < 4; ++g) {
      const bf16_t* sp = &bsm[cur][g][lm][lh * 16];
      v8bf lo = *(const v8bf*)sp;          // ds_load_b128
      v8bf hi = *(const v8bf*)(sp + 8);
      acc[g] = wmma_bf16(av, cat16(lo, hi), acc[g]);
    }
  }
}

// -------------------- recurrent step kernels --------------------------------
// Block = one 16-col hidden tile (ht = blockIdx.x), wave w = batch tile.
__global__ void __launch_bounds__(128)
k_step0(const bf16_t* __restrict__ hin, const bf16_t* __restrict__ Ut,
        const float* __restrict__ P0, int t,
        float* __restrict__ c, bf16_t* __restrict__ hout,
        float* __restrict__ hf) {
  __shared__ __align__(16) bf16_t bsm[2][4][16][32];   // 8KB double buffer
  const int lane = threadIdx.x & 31;
  const int w    = threadIdx.x >> 5;
  const int row0 = w * 16;
  const int col0 = blockIdx.x * 16;
  const int lm = lane & 15, lh = lane >> 4;
  const unsigned ldsBase = (unsigned)(unsigned long long)(const void*)&bsm[0][0][0][0];

  v8f acc[4];
#pragma unroll
  for (int g = 0; g < 4; ++g) {
    int col = g * H_DIM + col0 + lm;
#pragma unroll
    for (int r = 0; r < 8; ++r) {
      int b = row0 + lh * 8 + r;
      acc[g][r] = P0[((size_t)b * S_LEN + t) * N4 + col];
    }
  }
  gemm_phase(hin, Ut, bsm, ldsBase, row0, col0, w, lane, acc);
#pragma unroll
  for (int r = 0; r < 8; ++r) {
    int b = row0 + lh * 8 + r;
    size_t idx = (size_t)b * H_DIM + col0 + lm;
    float fg = sigm(acc[0][r]);
    float ig = sigm(acc[1][r]);
    float gg = tanhf(acc[2][r]);
    float og = sigm(acc[3][r]);
    float cn = fg * c[idx] + ig * gg;
    float hn = og * tanhf(cn);
    c[idx]    = cn;
    hf[idx]   = hn;
    hout[idx] = f2bf(hn);
  }
}

__global__ void __launch_bounds__(128)
k_step1(const bf16_t* __restrict__ h0new, const bf16_t* __restrict__ W1t,
        const bf16_t* __restrict__ h1in,  const bf16_t* __restrict__ U1t,
        const float* __restrict__ P1, int t,
        float* __restrict__ c, bf16_t* __restrict__ hout,
        float* __restrict__ h1all) {
  __shared__ __align__(16) bf16_t bsm[2][4][16][32];
  const int lane = threadIdx.x & 31;
  const int w    = threadIdx.x >> 5;
  const int row0 = w * 16;
  const int col0 = blockIdx.x * 16;
  const int lm = lane & 15, lh = lane >> 4;
  const unsigned ldsBase = (unsigned)(unsigned long long)(const void*)&bsm[0][0][0][0];

  v8f acc[4];
#pragma unroll
  for (int g = 0; g < 4; ++g) {
    int col = g * H_DIM + col0 + lm;
#pragma unroll
    for (int r = 0; r < 8; ++r) {
      int b = row0 + lh * 8 + r;
      acc[g][r] = P1[((size_t)b * S_LEN + t) * N4 + col];
    }
  }
  gemm_phase(h0new, W1t, bsm, ldsBase, row0, col0, w, lane, acc);
  gemm_phase(h1in,  U1t, bsm, ldsBase, row0, col0, w, lane, acc);
#pragma unroll
  for (int r = 0; r < 8; ++r) {
    int b = row0 + lh * 8 + r;
    int col = col0 + lm;
    size_t idx = (size_t)b * H_DIM + col;
    float fg = sigm(acc[0][r]);
    float ig = sigm(acc[1][r]);
    float gg = tanhf(acc[2][r]);
    float og = sigm(acc[3][r]);
    float cn = fg * c[idx] + ig * gg;
    float hn = og * tanhf(cn);
    c[idx]    = cn;
    hout[idx] = f2bf(hn);
    h1all[((size_t)t * B_SZ + b) * H_DIM + col] = hn;
  }
}

// -------------------- output head + final states ---------------------------
__global__ void __launch_bounds__(128)
k_head(const float* __restrict__ h1all, const float* __restrict__ fcW,
       const float* __restrict__ fcb, float* __restrict__ y) {
  int lane = threadIdx.x & 31;
  int widx = blockIdx.x * 4 + (threadIdx.x >> 5);
  int b = widx >> 9;
  int s = widx & 511;
  const float* hp = h1all + ((size_t)s * B_SZ + b) * H_DIM;
  float s0 = 0.f, s1 = 0.f, s2 = 0.f;
  for (int h = lane; h < H_DIM; h += 32) {
    float v = hp[h];
    s0 += v * fcW[h * 3 + 0];
    s1 += v * fcW[h * 3 + 1];
    s2 += v * fcW[h * 3 + 2];
  }
#pragma unroll
  for (int off = 16; off > 0; off >>= 1) {
    s0 += __shfl_xor(s0, off, 32);
    s1 += __shfl_xor(s1, off, 32);
    s2 += __shfl_xor(s2, off, 32);
  }
  if (lane == 0) {
    size_t o = ((size_t)b * S_LEN + s) * OUTD;
    y[o + 0] = s0 + fcb[0];
    y[o + 1] = s1 + fcb[1];
    y[o + 2] = s2 + fcb[2];
  }
}

__global__ void k_final(const float* __restrict__ h0f, const float* __restrict__ h1all,
                        const float* __restrict__ c0, const float* __restrict__ c1,
                        float* __restrict__ out) {
  int i = blockIdx.x * blockDim.x + threadIdx.x;
  const int NBH = B_SZ * H_DIM;
  if (i >= NBH) return;
  const size_t yoff = (size_t)B_SZ * S_LEN * OUTD;
  out[yoff + i]           = h0f[i];
  out[yoff + NBH + i]     = h1all[(size_t)(S_LEN - 1) * NBH + i];
  out[yoff + 2 * NBH + i] = c0[i];
  out[yoff + 3 * NBH + i] = c1[i];
}

// ---------------------------------------------------------------------------
extern "C" void kernel_launch(void* const* d_in, const int* in_sizes, int n_in,
                              void* d_out, int out_size, void* d_ws, size_t ws_size,
                              hipStream_t stream) {
  const float* x   = (const float*)d_in[0];
  const float* z   = (const float*)d_in[1];
  const float* W0  = (const float*)d_in[2];
  const float* bW0 = (const float*)d_in[3];
  const float* U0  = (const float*)d_in[4];
  const float* V0  = (const float*)d_in[5];
  const float* b0  = (const float*)d_in[6];
  const float* W1  = (const float*)d_in[7];
  const float* bW1 = (const float*)d_in[8];
  const float* U1  = (const float*)d_in[9];
  const float* V1  = (const float*)d_in[10];
  const float* b1  = (const float*)d_in[11];
  const float* fcW = (const float*)d_in[12];
  const float* fcb = (const float*)d_in[13];
  float* out = (float*)d_out;

  char* ws = (char*)d_ws;
  size_t off = 0;
  auto carve = [&](size_t bytes) -> char* {
    char* p = ws + off;
    off += (bytes + 255) & ~(size_t)255;
    return p;
  };
  bf16_t* W0t = (bf16_t*)carve((size_t)DIN  * N4 * 2);
  bf16_t* U0t = (bf16_t*)carve((size_t)H_DIM * N4 * 2);
  bf16_t* V0t = (bf16_t*)carve((size_t)DADD * N4 * 2);
  bf16_t* W1t = (bf16_t*)carve((size_t)H_DIM * N4 * 2);
  bf16_t* U1t = (bf16_t*)carve((size_t)H_DIM * N4 * 2);
  bf16_t* V1t = (bf16_t*)carve((size_t)DADD * N4 * 2);
  bf16_t* xbf = (bf16_t*)carve((size_t)MTOT * DIN  * 2);
  bf16_t* zbf = (bf16_t*)carve((size_t)MTOT * DADD * 2);
  float*  P0  = (float*) carve((size_t)MTOT * N4 * 4);
  float*  P1  = (float*) carve((size_t)MTOT * N4 * 4);
  bf16_t* h0a = (bf16_t*)carve((size_t)B_SZ * H_DIM * 2);
  bf16_t* h0b = (bf16_t*)carve((size_t)B_SZ * H_DIM * 2);
  bf16_t* h1a = (bf16_t*)carve((size_t)B_SZ * H_DIM * 2);
  bf16_t* h1b = (bf16_t*)carve((size_t)B_SZ * H_DIM * 2);
  float*  c0s = (float*) carve((size_t)B_SZ * H_DIM * 4);
  float*  c1s = (float*) carve((size_t)B_SZ * H_DIM * 4);
  float*  h0f = (float*) carve((size_t)B_SZ * H_DIM * 4);
  float*  h1all = (float*)carve((size_t)S_LEN * B_SZ * H_DIM * 4);
  (void)ws_size; (void)in_sizes; (void)n_in; (void)out_size;

  // ---- weight/input conversion ----
  k_cvt_t<<<dim3(N4 / 32, DIN   / 32), 256, 0, stream>>>(W0, W0t, DIN,  N4);
  k_cvt_t<<<dim3(N4 / 32, H_DIM / 32), 256, 0, stream>>>(U0, U0t, H_DIM, N4);
  k_cvt_t<<<dim3(N4 / 32, DADD  / 32), 256, 0, stream>>>(V0, V0t, DADD, N4);
  k_cvt_t<<<dim3(N4 / 32, H_DIM / 32), 256, 0, stream>>>(W1, W1t, H_DIM, N4);
  k_cvt_t<<<dim3(N4 / 32, H_DIM / 32), 256, 0, stream>>>(U1, U1t, H_DIM, N4);
  k_cvt_t<<<dim3(N4 / 32, DADD  / 32), 256, 0, stream>>>(V1, V1t, DADD, N4);
  {
    size_t nx = (size_t)MTOT * DIN, nz = (size_t)MTOT * DADD;
    k_cvt<<<(unsigned)((nx + 255) / 256), 256, 0, stream>>>(x, xbf, nx);
    k_cvt<<<(unsigned)((nz + 255) / 256), 256, 0, stream>>>(z, zbf, nz);
  }

  // ---- zero initial recurrent state ----
  hipMemsetAsync(h0a, 0, (size_t)B_SZ * H_DIM * 2, stream);
  hipMemsetAsync(h1a, 0, (size_t)B_SZ * H_DIM * 2, stream);
  hipMemsetAsync(c0s, 0, (size_t)B_SZ * H_DIM * 4, stream);
  hipMemsetAsync(c1s, 0, (size_t)B_SZ * H_DIM * 4, stream);

  // ---- time-parallel projections ----
  const unsigned preBlocks = (MTOT / 16) * (N4 / 64) / 4;   // 32768
  k_precompute<<<preBlocks, 128, 0, stream>>>(xbf, DIN, DIN, W0t,
                                              zbf, DADD, DADD, V0t,
                                              bW0, b0, P0);
  k_precompute<<<preBlocks, 128, 0, stream>>>(zbf, DADD, DADD, V1t,
                                              nullptr, 0, 0, nullptr,
                                              bW1, b1, P1);

  // ---- sequential recurrence: 512 steps, ping-pong h buffers ----
  for (int t = 0; t < S_LEN; ++t) {
    const bf16_t* h0i = (t & 1) ? h0b : h0a;
    bf16_t*       h0o = (t & 1) ? h0a : h0b;
    const bf16_t* h1i = (t & 1) ? h1b : h1a;
    bf16_t*       h1o = (t & 1) ? h1a : h1b;
    k_step0<<<H_DIM / 16, 128, 0, stream>>>(h0i, U0t, P0, t, c0s, h0o, h0f);
    k_step1<<<H_DIM / 16, 128, 0, stream>>>(h0o, W1t, h1i, U1t, P1, t, c1s, h1o, h1all);
  }

  // ---- output head + final states ----
  k_head<<<(B_SZ * S_LEN) / 4, 128, 0, stream>>>(h1all, fcW, fcb, out);
  k_final<<<(B_SZ * H_DIM + 255) / 256, 256, 0, stream>>>(h0f, h1all, c0s, c1s, out);
}
